// GCN_model_3977139716685
// MI455X (gfx1250) — matrix operand order repaired and checked by reading, not compiled, not run on hardware
//
#include <hip/hip_runtime.h>
#include <hip/hip_bf16.h>
#include <cstdint>

#define C_DIM 512
#define N_PTS 4096
#define B_DIM 4
#define M_TOT (B_DIM * N_PTS)   // 16384 rows
#define KNN_K 5
#define EPS_BN 1e-5f

typedef __attribute__((ext_vector_type(16))) __bf16 v16bf;
typedef __attribute__((ext_vector_type(8)))  float  v8f;

union ABfrag { unsigned int u[8]; v16bf v; };

static __device__ __forceinline__ unsigned int f2bf(float f) {
  unsigned int u = __float_as_uint(f);
  return (u + 0x7FFFu + ((u >> 16) & 1u)) >> 16;   // round-to-nearest-even
}
static __device__ __forceinline__ unsigned int pack2(float lo, float hi) {
  return f2bf(lo) | (f2bf(hi) << 16);
}
// low 32 bits of a generic LDS pointer == DS-relative byte address (aperture truncation)
static __device__ __forceinline__ unsigned int lds_addr32(const void* p) {
  return (unsigned int)(unsigned long long)(size_t)p;
}

// ------- one-shot W[Nout,K] fp32 -> Wt[K,Nout] bf16 (LDS-tiled transpose) -------
__global__ void transpose_w_bf16(const float* __restrict__ W, unsigned short* __restrict__ Wt,
                                 int Nout, int Kdim) {
  __shared__ float tile[32][33];
  const int kb = blockIdx.x * 32;        // K tile
  const int nb = blockIdx.y * 32;        // N tile
  const int tx = threadIdx.x, ty = threadIdx.y;   // 32 x 8
#pragma unroll
  for (int j = 0; j < 32; j += 8) {
    int n = nb + ty + j;
    tile[ty + j][tx] = (n < Nout) ? W[(size_t)n * Kdim + kb + tx] : 0.0f;
  }
  __syncthreads();
#pragma unroll
  for (int j = 0; j < 32; j += 8) {
    int k = kb + ty + j;
    int n = nb + tx;
    if (n < Nout)
      Wt[(size_t)k * Nout + n] = (unsigned short)f2bf(tile[tx][ty + j]);
  }
}

// ---------------- [B,C,N] -> [B,N,C] transpose (LDS tiled) ----------------
__global__ void transpose_cn(const float* __restrict__ x, float* __restrict__ xt) {
  __shared__ float tile[32][33];
  const int nb = blockIdx.x * 32, cb = blockIdx.y * 32, b = blockIdx.z;
  const int tx = threadIdx.x, ty = threadIdx.y;
#pragma unroll
  for (int k = 0; k < 32; k += 8)
    tile[ty + k][tx] = x[(size_t)(b * C_DIM + cb + ty + k) * N_PTS + nb + tx];
  __syncthreads();
#pragma unroll
  for (int k = 0; k < 32; k += 8)
    xt[(size_t)(b * N_PTS + nb + ty + k) * C_DIM + cb + tx] = tile[tx][ty + k];
}

// ---------------- brute-force KNN (k=5, incl. self) ----------------
__global__ void knn_kernel(const float* __restrict__ pos, int* __restrict__ idx) {
  __shared__ float2 tile[256];
  const int gi = blockIdx.x * 256 + threadIdx.x;
  const int b = gi / N_PTS;
  const float2* pts = ((const float2*)pos) + (size_t)b * N_PTS;
  const float2 q = ((const float2*)pos)[gi];
  float bd[KNN_K]; int bi[KNN_K];
#pragma unroll
  for (int t = 0; t < KNN_K; ++t) { bd[t] = 3.4e38f; bi[t] = 0; }
  for (int jt = 0; jt < N_PTS; jt += 256) {
    __syncthreads();
    tile[threadIdx.x] = pts[jt + threadIdx.x];
    __syncthreads();
    for (int jj = 0; jj < 256; ++jj) {
      float dx = q.x - tile[jj].x, dy = q.y - tile[jj].y;
      float d2 = dx * dx + dy * dy;
      if (d2 < bd[KNN_K - 1]) {           // strict < : earliest index wins ties (matches top_k)
        int j = jt + jj;
        int p = KNN_K - 1;
        while (p > 0 && d2 < bd[p - 1]) { bd[p] = bd[p - 1]; bi[p] = bi[p - 1]; --p; }
        bd[p] = d2; bi[p] = j;
      }
    }
  }
#pragma unroll
  for (int t = 0; t < KNN_K; ++t) idx[(size_t)gi * KNN_K + t] = bi[t];
}

// ------- gather + max-relative; writes feat directly as packed bf16 -------
__global__ void gather_maxrel(const float* __restrict__ xt, const int* __restrict__ idx,
                              unsigned int* __restrict__ featbf) {
  const int p = blockIdx.x;
  const int bbase = (p / N_PTS) * N_PTS;
  __shared__ int nb[KNN_K];
  if (threadIdx.x < KNN_K) nb[threadIdx.x] = idx[(size_t)p * KNN_K + threadIdx.x];
  __syncthreads();
  const float* self = xt + (size_t)p * C_DIM;
  const float* r0 = xt + (size_t)(bbase + nb[0]) * C_DIM;
  const float* r1 = xt + (size_t)(bbase + nb[1]) * C_DIM;
  const float* r2 = xt + (size_t)(bbase + nb[2]) * C_DIM;
  const float* r3 = xt + (size_t)(bbase + nb[3]) * C_DIM;
  const float* r4 = xt + (size_t)(bbase + nb[4]) * C_DIM;
  unsigned int* frow = featbf + (size_t)p * C_DIM;   // C_DIM uints = 2*C_DIM bf16
  for (int c = threadIdx.x; c < C_DIM; c += 256) {
    float s = self[c];
    float m = r0[c] - s;
    m = fmaxf(m, r1[c] - s);
    m = fmaxf(m, r2[c] - s);
    m = fmaxf(m, r3[c] - s);
    m = fmaxf(m, r4[c] - s);
    frow[c] = pack2(s, m);
  }
}

// ---------------- WMMA GEMM: Y[M,Nout] = A[M,K] * W^T + bias ----------------
// A bf16 row-major [M,K]; Wt bf16 K-major [K,Nout].
// block = 256 thr (8 waves); block tile 256(M) x 64(N); each wave: 2 M-tiles x 4 N-tiles.
// B tile staged global->LDS with async b128 copies (ASYNCcnt), double buffered.
// mode: 0 = f32 out, 1 = f32 out + relu, 2 = bf16 out + relu
__global__ void __launch_bounds__(256)
gemm_bf16(const unsigned short* __restrict__ A, const unsigned short* __restrict__ Wt,
          const float* __restrict__ bias, void* __restrict__ Yout,
          int Kdim, int Nout, int mode) {
  __shared__ unsigned short Bs[2][32][72];   // [k][n] tile; 144B row stride (16B aligned, padded)
  const int lane = threadIdx.x & 31;
  const int wave = threadIdx.x >> 5;
  const int half = lane >> 4;
  const int m0 = blockIdx.x * 256 + wave * 16;       // second M-tile at +128
  const int n0 = blockIdx.y * 64;
  const int mrow0 = m0 + (lane & 15);
  const int mrow1 = mrow0 + 128;
  const int sk = threadIdx.x >> 3;                   // staging: k row (0..31)
  const int sc = (threadIdx.x & 7) << 3;             // staging: n chunk (8 bf16 = 16B)
  v8f acc[2][4] = {};

  // async-stage one 32(K) x 64(N) bf16 tile of Wt into LDS buffer `buf`
  auto stage = [&](int buf, int k0) {
    const unsigned short* src = Wt + (size_t)(k0 + sk) * Nout + n0 + sc;
    unsigned int dst = lds_addr32(&Bs[buf][sk][sc]);
    asm volatile("global_load_async_to_lds_b128 %0, %1, off"
                 :: "v"(dst), "v"((unsigned long long)(size_t)src)
                 : "memory");
  };

  stage(0, 0);
  const int nsteps = Kdim >> 5;
  for (int s = 0; s < nsteps; ++s) {
    const int k0 = s << 5;
    const int buf = s & 1;
    asm volatile("s_wait_asynccnt 0x0" ::: "memory");  // my async copy landed in LDS
    __syncthreads();                                   // everyone's copy visible
    if (s + 1 < nsteps) stage(buf ^ 1, k0 + 32);

    // A fragments (row-major bf16): per lane two contiguous 16B chunks,
    // K = 8*half..+7 and K = 16+8*half..+7  (GLOBAL_LOAD_B128 path)
    const uint4* ap0 = (const uint4*)(A + (size_t)mrow0 * Kdim + k0 + 8 * half);
    const uint4* ap1 = (const uint4*)(A + (size_t)mrow1 * Kdim + k0 + 8 * half);
    __builtin_prefetch((const char*)ap0 + 64, 0, 3);
    __builtin_prefetch((const char*)ap1 + 64, 0, 3);
    uint4 a00 = ap0[0], a01 = ap0[2];
    uint4 a10 = ap1[0], a11 = ap1[2];
    ABfrag afr0, afr1;
    afr0.u[0] = a00.x; afr0.u[1] = a00.y; afr0.u[2] = a00.z; afr0.u[3] = a00.w;
    afr0.u[4] = a01.x; afr0.u[5] = a01.y; afr0.u[6] = a01.z; afr0.u[7] = a01.w;
    afr1.u[0] = a10.x; afr1.u[1] = a10.y; afr1.u[2] = a10.z; afr1.u[3] = a10.w;
    afr1.u[4] = a11.x; afr1.u[5] = a11.y; afr1.u[6] = a11.z; afr1.u[7] = a11.w;

    // 4 B fragments (lane holds column N=lane&15, VGPR v holds K=8*half+2v,+1 (+8 for v>=4)), shared by both M-tiles
#pragma unroll
    for (int nt = 0; nt < 4; ++nt) {
      ABfrag bfr;
      const int bn = nt * 16 + (lane & 15);
#pragma unroll
      for (int j = 0; j < 8; ++j) {
        const int kk = 8 * half + 2 * j + (j >= 4 ? 8 : 0);
        bfr.u[j] = (unsigned int)Bs[buf][kk][bn] | ((unsigned int)Bs[buf][kk + 1][bn] << 16);
      }
      acc[0][nt] = __builtin_amdgcn_wmma_f32_16x16x32_bf16(
          false, afr0.v, false, bfr.v, (short)0, acc[0][nt], false, false);
      acc[1][nt] = __builtin_amdgcn_wmma_f32_16x16x32_bf16(
          false, afr1.v, false, bfr.v, (short)0, acc[1][nt], false, false);
    }
  }

  // epilogue: 32-bit C/D layout — VGPR r -> M = r + 8*half, N = lane&15
  const int ncol = lane & 15;
#pragma unroll
  for (int t = 0; t < 2; ++t) {
#pragma unroll
    for (int nt = 0; nt < 4; ++nt) {
      int n = n0 + nt * 16 + ncol;
      float bia = bias[n];
#pragma unroll
      for (int r = 0; r < 8; ++r) {
        int m = m0 + t * 128 + r + 8 * half;
        float v = acc[t][nt][r] + bia;
        if (mode >= 1) v = fmaxf(v, 0.0f);
        if (mode == 2)
          ((unsigned short*)Yout)[(size_t)m * Nout + n] = (unsigned short)f2bf(v);
        else
          ((float*)Yout)[(size_t)m * Nout + n] = v;
      }
    }
  }
}

// ------------- final WMMA GEMM: out[M,5] = A[M,K] * Wc2^T + b (Wt2 is [K,5]) -------------
__global__ void __launch_bounds__(256)
gemm_final(const unsigned short* __restrict__ A, const unsigned short* __restrict__ W2t,
           const float* __restrict__ b2, float* __restrict__ out, int Kdim) {
  const int lane = threadIdx.x & 31;
  const int wave = threadIdx.x >> 5;
  const int half = lane >> 4;
  const int m0 = blockIdx.x * 128 + wave * 16;
  const int mrow = m0 + (lane & 15);
  v8f acc = {};
  for (int k0 = 0; k0 < Kdim; k0 += 32) {
    const uint4* ap = (const uint4*)(A + (size_t)mrow * Kdim + k0 + 8 * half);
    uint4 a0 = ap[0];
    uint4 a1 = ap[2];
    ABfrag afr, bfr;
    afr.u[0] = a0.x; afr.u[1] = a0.y; afr.u[2] = a0.z; afr.u[3] = a0.w;
    afr.u[4] = a1.x; afr.u[5] = a1.y; afr.u[6] = a1.z; afr.u[7] = a1.w;
    const int bn = lane & 15;   // B column N = lane&15, K in table order
#pragma unroll
    for (int j = 0; j < 8; ++j) {
      const int kk = k0 + 8 * half + 2 * j + (j >= 4 ? 8 : 0);
      unsigned int lo = (bn < 5) ? (unsigned int)W2t[(size_t)kk * 5 + bn] : 0u;
      unsigned int hi = (bn < 5) ? (unsigned int)W2t[(size_t)(kk + 1) * 5 + bn] : 0u;
      bfr.u[j] = lo | (hi << 16);
    }
    acc = __builtin_amdgcn_wmma_f32_16x16x32_bf16(
        false, afr.v, false, bfr.v, (short)0, acc, false, false);
  }
  const int n = lane & 15;
  if (n < 5) {                                // divergence only after last WMMA
    float bia = b2[n];
#pragma unroll
    for (int r = 0; r < 8; ++r) {
      int m = m0 + r + 8 * half;
      out[(size_t)m * 5 + n] = acc[r] + bia;
    }
  }
}

// --------- BN stage 1: coalesced per-row-block partial sums (64 blocks) ---------
__global__ void bn_part(const float* __restrict__ Y, float* __restrict__ psum,
                        float* __restrict__ psq) {
  const int rb = blockIdx.x * 256;
  const int c = threadIdx.x;                 // owns channels c and c+256
  float s0 = 0.f, q0 = 0.f, s1 = 0.f, q1 = 0.f;
  for (int r = 0; r < 256; ++r) {
    const float* row = Y + (size_t)(rb + r) * C_DIM;
    float a = row[c];
    float b = row[c + 256];
    s0 += a; q0 += a * a;
    s1 += b; q1 += b * b;
  }
  float* ps = psum + (size_t)blockIdx.x * C_DIM;
  float* pq = psq  + (size_t)blockIdx.x * C_DIM;
  ps[c] = s0; ps[c + 256] = s1;
  pq[c] = q0; pq[c + 256] = q1;
}

// --------- BN stage 2: finish 64 partials -> (scale, shift) per channel ---------
__global__ void bn_finish(const float* __restrict__ psum, const float* __restrict__ psq,
                          const float* __restrict__ g, const float* __restrict__ be,
                          float* __restrict__ stats) {
  const int c = blockIdx.x * 256 + threadIdx.x;
  float s = 0.f, q = 0.f;
  for (int b = 0; b < M_TOT / 256; ++b) {
    s += psum[(size_t)b * C_DIM + c];
    q += psq [(size_t)b * C_DIM + c];
  }
  float mean = s * (1.0f / (float)M_TOT);
  float var  = q * (1.0f / (float)M_TOT) - mean * mean;
  float scale = g[c] * rsqrtf(var + EPS_BN);
  stats[c]         = scale;
  stats[C_DIM + c] = be[c] - mean * scale;
}

// ------- normalize + ReLU in place (fp32) + packed-bf16 copy for classifier -------
__global__ void bn_relu(float* __restrict__ Y, const float* __restrict__ stats,
                        unsigned int* __restrict__ Ybf) {
  const int i = blockIdx.x * blockDim.x + threadIdx.x;   // float4 index, grid sized exactly
  float4 v = ((float4*)Y)[i];
  const int c = (i << 2) & (C_DIM - 1);
  v.x = fmaxf(v.x * stats[c + 0] + stats[C_DIM + c + 0], 0.0f);
  v.y = fmaxf(v.y * stats[c + 1] + stats[C_DIM + c + 1], 0.0f);
  v.z = fmaxf(v.z * stats[c + 2] + stats[C_DIM + c + 2], 0.0f);
  v.w = fmaxf(v.w * stats[c + 3] + stats[C_DIM + c + 3], 0.0f);
  ((float4*)Y)[i] = v;
  uint2 pb;
  pb.x = pack2(v.x, v.y);
  pb.y = pack2(v.z, v.w);
  ((uint2*)Ybf)[i] = pb;
}

extern "C" void kernel_launch(void* const* d_in, const int* in_sizes, int n_in,
                              void* d_out, int out_size, void* d_ws, size_t ws_size,
                              hipStream_t stream) {
  (void)in_sizes; (void)n_in; (void)out_size; (void)ws_size;
  const float* x   = (const float*)d_in[0];
  const float* pos = (const float*)d_in[1];
  const float* Wm[3]  = {(const float*)d_in[2],  (const float*)d_in[6],  (const float*)d_in[10]};
  const float* bm[3]  = {(const float*)d_in[3],  (const float*)d_in[7],  (const float*)d_in[11]};
  const float* gm[3]  = {(const float*)d_in[4],  (const float*)d_in[8],  (const float*)d_in[12]};
  const float* bem[3] = {(const float*)d_in[5],  (const float*)d_in[9],  (const float*)d_in[13]};
  const float* Wc1 = (const float*)d_in[14];
  const float* bc1 = (const float*)d_in[15];
  const float* Wc2 = (const float*)d_in[16];
  const float* bc2 = (const float*)d_in[17];
  float* out = (float*)d_out;

  // workspace carve-up (~130 MB total)
  char* ws = (char*)d_ws;
  size_t off = 0;
  auto carve = [&](size_t bytes) -> void* {
    void* p = ws + off;
    off = (off + bytes + 255) & ~(size_t)255;
    return p;
  };
  float*          bufA   = (float*)carve((size_t)M_TOT * C_DIM * 4);           // fp32 ping
  float*          bufY   = (float*)carve((size_t)M_TOT * C_DIM * 4);           // fp32 pong
  unsigned int*   featbf = (unsigned int*)carve((size_t)M_TOT * C_DIM * 4);    // [16384,1024] bf16
  unsigned int*   xtbf   = (unsigned int*)carve((size_t)M_TOT * C_DIM * 2);    // [16384,512] bf16
  unsigned short* hbf    = (unsigned short*)carve((size_t)M_TOT * (C_DIM/2) * 2);
  int*            idx    = (int*)carve((size_t)M_TOT * KNN_K * 4);
  float*          stats  = (float*)carve(2 * C_DIM * 4);
  float*          psum   = (float*)carve((size_t)(M_TOT / 256) * C_DIM * 4);
  float*          psq    = (float*)carve((size_t)(M_TOT / 256) * C_DIM * 4);
  unsigned short* wtbf[3];
  for (int i = 0; i < 3; ++i)
    wtbf[i] = (unsigned short*)carve((size_t)(2 * C_DIM) * C_DIM * 2);         // [1024,512]
  unsigned short* wc1t = (unsigned short*)carve((size_t)C_DIM * (C_DIM/2) * 2);// [512,256]
  unsigned short* wc2t = (unsigned short*)carve((size_t)(C_DIM/2) * 5 * 2);    // [256,5]

  // 0) one-shot weight transpose+convert to K-major bf16
  for (int i = 0; i < 3; ++i)
    transpose_w_bf16<<<dim3(2 * C_DIM / 32, C_DIM / 32), dim3(32, 8), 0, stream>>>(
        Wm[i], wtbf[i], C_DIM, 2 * C_DIM);
  transpose_w_bf16<<<dim3(C_DIM / 32, (C_DIM / 2) / 32), dim3(32, 8), 0, stream>>>(
      Wc1, wc1t, C_DIM / 2, C_DIM);
  transpose_w_bf16<<<dim3((C_DIM / 2) / 32, 1), dim3(32, 8), 0, stream>>>(
      Wc2, wc2t, 5, C_DIM / 2);
  // 1) x [B,C,H,W] -> xt [B*N, C]
  transpose_cn<<<dim3(N_PTS / 32, C_DIM / 32, B_DIM), dim3(32, 8), 0, stream>>>(x, bufA);
  // 2) KNN graph (same for all three blocks)
  knn_kernel<<<M_TOT / 256, 256, 0, stream>>>(pos, idx);
  // 3) three max-relative graph-conv blocks
  for (int blk = 0; blk < 3; ++blk) {
    gather_maxrel<<<M_TOT, 256, 0, stream>>>(bufA, idx, featbf);
    gemm_bf16<<<dim3(M_TOT / 256, C_DIM / 64), 256, 0, stream>>>(
        (const unsigned short*)featbf, wtbf[blk], bm[blk], bufY,
        2 * C_DIM, C_DIM, /*mode=*/0);
    bn_part<<<M_TOT / 256, 256, 0, stream>>>(bufY, psum, psq);
    bn_finish<<<C_DIM / 256, 256, 0, stream>>>(psum, psq, gm[blk], bem[blk], stats);
    bn_relu<<<(M_TOT * C_DIM / 4) / 256, 256, 0, stream>>>(bufY, stats, xtbf);
    float* t = bufA; bufA = bufY; bufY = t;
  }
  // 4) classifier: h = relu(xt @ Wc1^T + bc1) -> bf16 [16384, 256]
  gemm_bf16<<<dim3(M_TOT / 256, (C_DIM / 2) / 64), 256, 0, stream>>>(
      (const unsigned short*)xtbf, wc1t, bc1, hbf, C_DIM, C_DIM / 2, /*mode=*/2);
  // 5) out = h @ Wc2^T + bc2   [16384, 5]
  gemm_final<<<M_TOT / 128, 256, 0, stream>>>(hbf, wc2t, bc2, out, C_DIM / 2);
}
